// GIBModel_687194768135
// MI455X (gfx1250) — matrix operand order repaired and checked by reading, not compiled, not run on hardware
//
#include <hip/hip_runtime.h>
#include <hip/hip_bf16.h>

// ---------------- problem constants ----------------
#define NND   51200      // nodes
#define NG    256        // graphs
#define NPG   200        // nodes per graph
#define NE    409600     // intra-graph edges
#define NET   (NE + NND) // edges + self loops
#define FIN   768
#define HIDC  128
#define NHEAD 4
#define F1    (NHEAD * HIDC)   // 512

// ---------------- WMMA types ----------------
typedef __attribute__((ext_vector_type(16))) __bf16 v16bf;
typedef __attribute__((ext_vector_type(8)))  float  v8f;

union FragAB { v16bf v; uint4 q[2]; };
union FragC  { v8f  v; float f[8]; };

// ---------------- helpers ----------------
__device__ __forceinline__ unsigned short f2bf(float f) {
  unsigned u = __float_as_uint(f);
  return (unsigned short)((u + 0x7FFFu + ((u >> 16) & 1u)) >> 16);
}

__device__ __forceinline__ void atomicMaxFloat(float* addr, float val) {
  if (val >= 0.0f) atomicMax((int*)addr, __float_as_int(val));
  else             atomicMin((unsigned int*)addr, __float_as_uint(val));
}

// CDNA5 async memory->LDS copy, 16B per lane, tracked by ASYNCcnt.
__device__ __forceinline__ void async_copy_b128(const unsigned short* g, const unsigned short* lds) {
  unsigned lds_off = (unsigned)(unsigned long long)(const void*)lds;   // low 32 bits = LDS offset
  asm volatile("global_load_async_to_lds_b128 %0, %1, off"
               :: "v"(lds_off), "v"((unsigned long long)(const void*)g)
               : "memory");
}
__device__ __forceinline__ void wait_async0() {
  asm volatile("s_wait_asynccnt 0x0" ::: "memory");
}

// ---------------- utility kernels ----------------
__global__ void fill_f32(float* p, float v, long long n) {
  long long i = (long long)blockIdx.x * blockDim.x + threadIdx.x;
  if (i < n) p[i] = v;
}

__global__ void cvt_bf16(const float* __restrict__ in, unsigned short* __restrict__ out, long long n) {
  long long i = (long long)blockIdx.x * blockDim.x + threadIdx.x;
  if (i < n) out[i] = f2bf(in[i]);
}

// convert + transpose: in [K,Nc] f32 -> out [Nc,K] bf16 (weights, tiny)
__global__ void cvt_transpose_bf16(const float* __restrict__ in, unsigned short* __restrict__ out,
                                   int K, int Nc) {
  int idx = blockIdx.x * blockDim.x + threadIdx.x;
  if (idx >= K * Nc) return;
  int k = idx / Nc, n = idx % Nc;
  out[(size_t)n * K + k] = f2bf(in[idx]);
}

// ---------------- bf16 WMMA GEMM : C[M,Nc] = A[M,K] * B[K,Nc] ----------------
// A row-major bf16 [M,K]; Bt row-major bf16 [Nc,K] (B pre-transposed); C fp32.
// M % 128 == 0, Nc % 128 == 0, K % 32 == 0.
// 256 threads = 8 waves; block tile 128x128; wave tile 16x128 (8 WMMA / K-step).
// Double-buffered LDS filled by GLOBAL_LOAD_ASYNC_TO_LDS_B128.
#define BM 128
#define BN 128
#define BK 32

__global__ __launch_bounds__(256) void gemm_bf16_wmma(
    const unsigned short* __restrict__ A, const unsigned short* __restrict__ Bt,
    float* __restrict__ C, const float* __restrict__ bias, int Nc, int K, int relu) {
  __shared__ __align__(16) unsigned short As[2][BM * BK];   // [row][k]
  __shared__ __align__(16) unsigned short Bs[2][BN * BK];   // [n][k]  (rows of B^T)

  const int tid  = threadIdx.x;
  const int wave = tid >> 5;
  const int lane = tid & 31;
  const int hh   = lane >> 4;
  const int lr   = lane & 15;

  const unsigned short* Ablk = A  + (size_t)blockIdx.x * BM * K;
  const unsigned short* Bblk = Bt + (size_t)blockIdx.y * BN * K;

  // per-thread staging chunks (16B each): A rows have 4 chunks (32 bf16)
  const int cA0r = tid >> 2,          cA0k = (tid & 3) * 8;          // chunk tid
  const int cA1r = (tid + 256) >> 2,  cA1k = ((tid + 256) & 3) * 8;  // chunk tid+256

  auto issue_tile = [&](int kt, int buf) {
    const int kb = kt * BK;
    async_copy_b128(Ablk + (size_t)cA0r * K + kb + cA0k, &As[buf][cA0r * BK + cA0k]);
    async_copy_b128(Ablk + (size_t)cA1r * K + kb + cA1k, &As[buf][cA1r * BK + cA1k]);
    async_copy_b128(Bblk + (size_t)cA0r * K + kb + cA0k, &Bs[buf][cA0r * BK + cA0k]);
    async_copy_b128(Bblk + (size_t)cA1r * K + kb + cA1k, &Bs[buf][cA1r * BK + cA1k]);
  };

  FragC acc[8] = {};
  const int kT = K / BK;

  issue_tile(0, 0);

  for (int kt = 0; kt < kT; ++kt) {
    const int buf = kt & 1;
    wait_async0();                 // tile kt resident in LDS (this wave's share)
    __syncthreads();               // all waves' shares resident; prev compute done
    if (kt + 1 < kT) issue_tile(kt + 1, buf ^ 1);   // overlaps with compute below

    // A fragment: row = wave*16+lr ; K = {8h..8h+7} U {16+8h..+7}   (ISA 16-bit A layout)
    FragAB af;
    af.q[0] = *(const uint4*)&As[buf][(wave * 16 + lr) * BK + 8 * hh];
    af.q[1] = *(const uint4*)&As[buf][(wave * 16 + lr) * BK + 16 + 8 * hh];

#pragma unroll
    for (int ct = 0; ct < 8; ++ct) {
      // B fragment: col = ct*16+lr ; K = {8h..8h+7} U {16+8h..+7}   (matches A layout)
      FragAB bf;
      bf.q[0] = *(const uint4*)&Bs[buf][(ct * 16 + lr) * BK + 8 * hh];
      bf.q[1] = *(const uint4*)&Bs[buf][(ct * 16 + lr) * BK + 16 + 8 * hh];
      acc[ct].v = __builtin_amdgcn_wmma_f32_16x16x32_bf16(
          false, af.v, false, bf.v, (short)0, acc[ct].v, false, false);
    }
    __syncthreads();               // compute done before buf is refilled at kt+2
  }

  // ---- epilogue: C layout lanes0-15 M=r, lanes16-31 M=8+r, N=lane&15 ----
#pragma unroll
  for (int ct = 0; ct < 8; ++ct) {
    int gcol = blockIdx.y * BN + ct * 16 + lr;
    float b = bias ? bias[gcol] : 0.0f;
#pragma unroll
    for (int r = 0; r < 8; ++r) {
      int grow = blockIdx.x * BM + wave * 16 + hh * 8 + r;
      float v = acc[ct].f[r] + b;
      if (relu) v = fmaxf(v, 0.0f);
      C[(size_t)grow * Nc + gcol] = v;
    }
  }
}

// ---------------- GAT attention kernels (fp32 scalar path) ----------------
__global__ void attn_coeff(const float* __restrict__ hpre, const float* __restrict__ as,
                           const float* __restrict__ ad, float* __restrict__ osrc,
                           float* __restrict__ odst, int n, int H, int C) {
  int tid = blockIdx.x * blockDim.x + threadIdx.x;
  if (tid >= n * H) return;
  int i = tid / H, h = tid % H;
  const float* hp  = hpre + (size_t)i * H * C + (size_t)h * C;
  const float* asp = as + h * C;
  const float* adp = ad + h * C;
  float s = 0.f, d = 0.f;
  for (int c = 0; c < C; ++c) { float v = hp[c]; s += v * asp[c]; d += v * adp[c]; }
  osrc[tid] = s; odst[tid] = d;
}

__device__ __forceinline__ void edge_sd(const int* ei, int e, int E, int& s, int& d) {
  if (e < E) { s = ei[e]; d = ei[E + e]; } else { s = d = e - E; }
}

__global__ void edge_logits(const int* __restrict__ ei, const float* __restrict__ asrc,
                            const float* __restrict__ adst, float* __restrict__ logits,
                            float* __restrict__ maxb, int E, int Ntot, int H) {
  int tid = blockIdx.x * blockDim.x + threadIdx.x;
  if (tid >= (E + Ntot) * H) return;
  int e = tid / H, h = tid % H;
  int s, d; edge_sd(ei, e, E, s, d);
  float l = asrc[s * H + h] + adst[d * H + h];
  l = (l > 0.f) ? l : 0.2f * l;                // leaky_relu 0.2
  logits[tid] = l;
  atomicMaxFloat(&maxb[d * H + h], l);
}

__global__ void edge_exp(const int* __restrict__ ei, float* __restrict__ logits,
                         const float* __restrict__ maxb, float* __restrict__ sumb,
                         int E, int Ntot, int H) {
  int tid = blockIdx.x * blockDim.x + threadIdx.x;
  if (tid >= (E + Ntot) * H) return;
  int e = tid / H, h = tid % H;
  int s, d; edge_sd(ei, e, E, s, d);
  float ex = __expf(logits[tid] - maxb[d * H + h]);
  logits[tid] = ex;
  atomicAdd(&sumb[d * H + h], ex);
}

// one wave per edge; lanes stride the F = H*C channels
__global__ void edge_aggregate(const int* __restrict__ ei, const float* __restrict__ hpre,
                               const float* __restrict__ alpha, const float* __restrict__ sumb,
                               float* __restrict__ out, int E, int Ntot, int H, int C) {
  int widx = blockIdx.x * (blockDim.x >> 5) + (threadIdx.x >> 5);
  int lane = threadIdx.x & 31;
  if (widx >= E + Ntot) return;
  int s, d; edge_sd(ei, widx, E, s, d);
  int F = H * C;
  for (int f = lane; f < F; f += 32) {
    int h = f / C;
    float a   = alpha[(size_t)widx * H + h];
    float den = sumb[d * H + h] + 1e-16f;
    atomicAdd(&out[(size_t)d * F + f], (a / den) * hpre[(size_t)s * F + f]);
  }
}

__global__ void bias_relu(float* __restrict__ a, const float* __restrict__ b, long long n, int F) {
  long long i = (long long)blockIdx.x * blockDim.x + threadIdx.x;
  if (i < n) a[i] = fmaxf(a[i] + b[i % F], 0.0f);
}

// ---------------- pooling head ----------------
__global__ void classify_S(const float* __restrict__ h2, const float* __restrict__ Wc,
                           const float* __restrict__ bc, float* __restrict__ S, int n) {
  int i = blockIdx.x * blockDim.x + threadIdx.x;
  if (i >= n) return;
  const float* hp = h2 + (size_t)i * HIDC;
  float l0 = bc[0], l1 = bc[1];
  for (int c = 0; c < HIDC; ++c) { float h = hp[c]; l0 += h * Wc[2 * c]; l1 += h * Wc[2 * c + 1]; }
  float m = fmaxf(l0, l1);
  float e0 = __expf(l0 - m), e1 = __expf(l1 - m);
  float sm = e0 + e1;
  S[2 * i] = e0 / sm; S[2 * i + 1] = e1 / sm;
}

__global__ void agg_outer(const int* __restrict__ ei, const int* __restrict__ batch,
                          const float* __restrict__ S, float* __restrict__ agg, int E) {
  int e = blockIdx.x * blockDim.x + threadIdx.x;
  if (e >= E) return;
  int s = ei[e], d = ei[E + e];
  int g = batch[s];
  float s0 = S[2 * s], s1 = S[2 * s + 1], t0 = S[2 * d], t1 = S[2 * d + 1];
  atomicAdd(&agg[g * 4 + 0], s0 * t0);
  atomicAdd(&agg[g * 4 + 1], s0 * t1);
  atomicAdd(&agg[g * 4 + 2], s1 * t0);
  atomicAdd(&agg[g * 4 + 3], s1 * t1);
}

__global__ void agg_loss(const float* __restrict__ agg, float* __restrict__ loss, int G) {
  int g = blockIdx.x * blockDim.x + threadIdx.x;
  if (g >= G) return;
  float a00 = agg[g * 4], a01 = agg[g * 4 + 1], a10 = agg[g * 4 + 2], a11 = agg[g * 4 + 3];
  float r0 = fmaxf(fabsf(a00) + fabsf(a01), 1e-5f);
  float r1 = fmaxf(fabsf(a10) + fabsf(a11), 1e-5f);
  float d0 = a00 / r0 - 1.0f, d1 = a11 / r1 - 1.0f;
  atomicAdd(loss, 0.5f * (d0 * d0 + d1 * d1) / (float)G);
}

__global__ void sub_embed(const float* __restrict__ S, const float* __restrict__ h2,
                          float* __restrict__ sub) {
  int g = blockIdx.x, c = threadIdx.x;
  float acc = 0.f;
  int n0 = g * NPG;
  for (int i = 0; i < NPG; ++i)
    acc += S[2 * (n0 + i)] * h2[(size_t)(n0 + i) * HIDC + c];
  sub[g * HIDC + c] = acc;
}

__global__ void head_mlp(const float* __restrict__ sub, const float* __restrict__ Wf1,
                         const float* __restrict__ bf1, const float* __restrict__ Wf2,
                         const float* __restrict__ bf2, float* __restrict__ out) {
  __shared__ float ss[HIDC], st[HIDC];
  int g = blockIdx.x, c = threadIdx.x;
  ss[c] = sub[g * HIDC + c];
  __syncthreads();
  float acc = bf1[c];
  for (int j = 0; j < HIDC; ++j) acc += ss[j] * Wf1[j * HIDC + c];
  st[c] = fmaxf(acc, 0.0f);
  __syncthreads();
  if (c < 2) {
    float o = bf2[c];
    for (int j = 0; j < HIDC; ++j) o += st[j] * Wf2[j * 2 + c];
    out[g * 2 + c] = o;
  }
}

// ---------------- launcher ----------------
static inline dim3 grd(long long n) { return dim3((unsigned)((n + 255) / 256)); }

extern "C" void kernel_launch(void* const* d_in, const int* in_sizes, int n_in,
                              void* d_out, int out_size, void* d_ws, size_t ws_size,
                              hipStream_t stream) {
  const float* x    = (const float*)d_in[0];
  const int*   ei   = (const int*)d_in[1];
  const int*   batch= (const int*)d_in[2];
  const float* W1   = (const float*)d_in[3];
  const float* a_s1 = (const float*)d_in[4];
  const float* a_d1 = (const float*)d_in[5];
  const float* b1   = (const float*)d_in[6];
  const float* W2   = (const float*)d_in[7];
  const float* a_s2 = (const float*)d_in[8];
  const float* a_d2 = (const float*)d_in[9];
  const float* b2   = (const float*)d_in[10];
  const float* Wc   = (const float*)d_in[11];
  const float* bc   = (const float*)d_in[12];
  const float* Wf1  = (const float*)d_in[13];
  const float* bf1  = (const float*)d_in[14];
  const float* Wf2  = (const float*)d_in[15];
  const float* bf2  = (const float*)d_in[16];
  float* out = (float*)d_out;

  // ---- workspace carve ----
  char* p = (char*)d_ws;
  auto alloc = [&p](size_t b) -> void* { void* r = (void*)p; p += (b + 255) & ~(size_t)255; return r; };
  unsigned short* xb    = (unsigned short*)alloc((size_t)NND * FIN * 2);
  unsigned short* w1bt  = (unsigned short*)alloc((size_t)FIN * F1 * 2);   // [F1][FIN]
  unsigned short* w2bt  = (unsigned short*)alloc((size_t)F1 * HIDC * 2);  // [HIDC][F1]
  float* hpre1  = (float*)alloc((size_t)NND * F1 * 4);
  float* asrc1  = (float*)alloc((size_t)NND * NHEAD * 4);
  float* adst1  = (float*)alloc((size_t)NND * NHEAD * 4);
  float* logit1 = (float*)alloc((size_t)NET * NHEAD * 4);
  float* maxb1  = (float*)alloc((size_t)NND * NHEAD * 4);
  float* sumb1  = (float*)alloc((size_t)NND * NHEAD * 4);
  float* h1     = (float*)alloc((size_t)NND * F1 * 4);
  unsigned short* h1b = (unsigned short*)alloc((size_t)NND * F1 * 2);
  float* hpre2  = (float*)alloc((size_t)NND * HIDC * 4);
  float* asrc2  = (float*)alloc((size_t)NND * 4);
  float* adst2  = (float*)alloc((size_t)NND * 4);
  float* logit2 = (float*)alloc((size_t)NET * 4);
  float* maxb2  = (float*)alloc((size_t)NND * 4);
  float* sumb2  = (float*)alloc((size_t)NND * 4);
  float* h2     = (float*)alloc((size_t)NND * HIDC * 4);
  float* S      = (float*)alloc((size_t)NND * 2 * 4);
  float* agg    = (float*)alloc((size_t)NG * 4 * 4);
  float* sub    = (float*)alloc((size_t)NG * HIDC * 4);

  // ---- bf16 conversions (weights transposed for async row-contiguous staging) ----
  cvt_bf16<<<grd((long long)NND * FIN), 256, 0, stream>>>(x, xb, (long long)NND * FIN);
  cvt_transpose_bf16<<<grd((long long)FIN * F1),  256, 0, stream>>>(W1, w1bt, FIN, F1);
  cvt_transpose_bf16<<<grd((long long)F1 * HIDC), 256, 0, stream>>>(W2, w2bt, F1, HIDC);

  // ---- GAT layer 1 ----
  gemm_bf16_wmma<<<dim3(NND / BM, F1 / BN), 256, 0, stream>>>(xb, w1bt, hpre1, nullptr, F1, FIN, 0);
  attn_coeff<<<grd((long long)NND * NHEAD), 256, 0, stream>>>(hpre1, a_s1, a_d1, asrc1, adst1, NND, NHEAD, HIDC);
  fill_f32<<<grd((long long)NND * NHEAD), 256, 0, stream>>>(maxb1, -1e30f, (long long)NND * NHEAD);
  fill_f32<<<grd((long long)NND * NHEAD), 256, 0, stream>>>(sumb1, 0.0f,   (long long)NND * NHEAD);
  fill_f32<<<grd((long long)NND * F1),    256, 0, stream>>>(h1,    0.0f,   (long long)NND * F1);
  edge_logits<<<grd((long long)NET * NHEAD), 256, 0, stream>>>(ei, asrc1, adst1, logit1, maxb1, NE, NND, NHEAD);
  edge_exp<<<grd((long long)NET * NHEAD), 256, 0, stream>>>(ei, logit1, maxb1, sumb1, NE, NND, NHEAD);
  edge_aggregate<<<dim3((NET + 7) / 8), 256, 0, stream>>>(ei, hpre1, logit1, sumb1, h1, NE, NND, NHEAD, HIDC);
  bias_relu<<<grd((long long)NND * F1), 256, 0, stream>>>(h1, b1, (long long)NND * F1, F1);

  // ---- GAT layer 2 ----
  cvt_bf16<<<grd((long long)NND * F1), 256, 0, stream>>>(h1, h1b, (long long)NND * F1);
  gemm_bf16_wmma<<<dim3(NND / BM, HIDC / BN), 256, 0, stream>>>(h1b, w2bt, hpre2, nullptr, HIDC, F1, 0);
  attn_coeff<<<grd(NND), 256, 0, stream>>>(hpre2, a_s2, a_d2, asrc2, adst2, NND, 1, HIDC);
  fill_f32<<<grd(NND), 256, 0, stream>>>(maxb2, -1e30f, NND);
  fill_f32<<<grd(NND), 256, 0, stream>>>(sumb2, 0.0f,   NND);
  fill_f32<<<grd((long long)NND * HIDC), 256, 0, stream>>>(h2, 0.0f, (long long)NND * HIDC);
  edge_logits<<<grd(NET), 256, 0, stream>>>(ei, asrc2, adst2, logit2, maxb2, NE, NND, 1);
  edge_exp<<<grd(NET), 256, 0, stream>>>(ei, logit2, maxb2, sumb2, NE, NND, 1);
  edge_aggregate<<<dim3((NET + 7) / 8), 256, 0, stream>>>(ei, hpre2, logit2, sumb2, h2, NE, NND, 1, HIDC);
  bias_relu<<<grd((long long)NND * HIDC), 256, 0, stream>>>(h2, b2, (long long)NND * HIDC, HIDC);

  // ---- pooling / loss / head ----
  classify_S<<<grd(NND), 256, 0, stream>>>(h2, Wc, bc, S, NND);
  fill_f32<<<grd(NG * 4), 256, 0, stream>>>(agg, 0.0f, NG * 4);
  fill_f32<<<1, 1, 0, stream>>>(out + NG * 2, 0.0f, 1);
  agg_outer<<<grd(NE), 256, 0, stream>>>(ei, batch, S, agg, NE);
  agg_loss<<<grd(NG), 256, 0, stream>>>(agg, out + NG * 2, NG);
  sub_embed<<<dim3(NG), 128, 0, stream>>>(S, h2, sub);
  head_mlp<<<dim3(NG), 128, 0, stream>>>(sub, Wf1, bf1, Wf2, bf2, out);
}
